// StdGroupedQueryAttention_75814762709712
// MI455X (gfx1250) — matrix operand order repaired and checked by reading, not compiled, not run on hardware
//
#include <hip/hip_runtime.h>
#include <stdint.h>

// ---------------------------------------------------------------------------
// Problem constants (from reference): B=2, S=2048, D_MODEL=4096, N_HEAD=32,
// D_HEAD=128, N_KV_GROUPS=8, KV_DIM=1024, H_PER_G=4.
// ---------------------------------------------------------------------------
#define BATCH    2
#define SEQ      2048
#define DMODEL   4096
#define KVDIM    1024
#define DHEAD    128

typedef __attribute__((ext_vector_type(16))) __bf16 v16bf;
typedef __attribute__((ext_vector_type(8)))  float  v8f;

union FragAU { v16bf bf; uint32_t u[8]; };
union FragBU { v16bf bf; uint4    q[2]; };

__device__ __forceinline__ uint16_t f32_to_bf16_rne(float f) {
  uint32_t x = __builtin_bit_cast(uint32_t, f);
  uint32_t r = x + 0x7FFFu + ((x >> 16) & 1u);   // round-to-nearest-even
  return (uint16_t)(r >> 16);
}
__device__ __forceinline__ uint32_t pack_bf16x2(float lo, float hi) {
  return (uint32_t)f32_to_bf16_rne(lo) | ((uint32_t)f32_to_bf16_rne(hi) << 16);
}

// D = A(16x32) * B(32x16) + C, bf16 in / f32 accumulate
__device__ __forceinline__ v8f wmma_bf16(v16bf a, v16bf b, v8f c) {
  return __builtin_amdgcn_wmma_f32_16x16x32_bf16(
      /*neg_a=*/false, a, /*neg_b=*/false, b,
      /*c_mod=*/(short)0, c, /*reuse_a=*/false, /*reuse_b=*/false);
}

// --- CDNA5 async global->LDS copy (ASYNCcnt), 16 bytes per lane ------------
// LDS destination byte offset taken from lds_off VGPR; global addr from VGPR pair.
__device__ __forceinline__ void async_ld_b128(uint32_t lds_off,
                                              const void* gptr) {
  asm volatile("global_load_async_to_lds_b128 %0, %1, off"
               :: "v"(lds_off), "v"(gptr)
               : "memory");
}
__device__ __forceinline__ void wait_async0() {
  asm volatile("s_wait_asynccnt 0" ::: "memory");
}
// Generic pointer to LDS: aperture lives in the high word, low 32 bits are
// the wave-relative LDS byte offset (ISA aperture-check equation).
__device__ __forceinline__ uint32_t lds_off_of(const void* p) {
  return (uint32_t)(size_t)p;
}

// A-matrix fragment (16x32 bf16, M x K), source row-major, K contiguous.
// Lane L: row = L&15;  element i -> K = (i<8 ? i : i+8) + 8*(L>>4).
__device__ __forceinline__ v16bf load_frag_a(const uint16_t* __restrict__ base,
                                             int ld) {
  const int lane = threadIdx.x & 31;
  const uint16_t* rp = base + (size_t)(lane & 15) * ld + ((lane >> 4) << 3);
  FragAU fa;
#pragma unroll
  for (int v = 0; v < 4; ++v)
    fa.u[v] = *(const uint32_t*)(rp + 2 * v);
#pragma unroll
  for (int v = 0; v < 4; ++v)
    fa.u[4 + v] = *(const uint32_t*)(rp + 16 + 2 * v);
  return fa.bf;
}

// B-matrix fragment (32x16 bf16, K x N), source stored [N][K] row-major
// (i.e. computes against the K-contiguous operand => C = A * B_src^T form).
// Lane L: col n = L&15; element i -> K = (i<8 ? i : i+8) + 8*(L>>4). K contiguous -> b128.
__device__ __forceinline__ v16bf load_frag_b(const uint16_t* __restrict__ base,
                                             int ld) {
  const int lane = threadIdx.x & 31;
  const uint16_t* p = base + (size_t)(lane & 15) * ld + ((lane >> 4) << 3);
  FragBU fb;
  fb.q[0] = ((const uint4*)p)[0];
  fb.q[1] = *(const uint4*)(p + 16);
  return fb.bf;
}

// ---------------------------------------------------------------------------
// GEMM: C[M,N] = A[M,K] * B[N,K]^T   (fp32 or bf16 A, fp32 B weights)
// 128x128 block tile, K-step 32. 256 threads = 8 waves; wave = 32x64 subtile
// (2 x 4 WMMA 16x16 tiles). LDS tiles padded to ld=40 (16B-aligned rows).
// fp32 operands are rounded to bf16 in-register while staging to LDS.
// ---------------------------------------------------------------------------
template <bool A_BF16, bool OUT_F32, bool TRANS_OUT>
__global__ __launch_bounds__(256) void gemm_bf16_wmma(
    const void* __restrict__ Av, const float* __restrict__ B,
    void* __restrict__ Cv, int M, int N, int K) {
  __shared__ __align__(16) uint16_t As[128 * 40];
  __shared__ __align__(16) uint16_t Bs[128 * 40];

  const int nbase = blockIdx.x * 128;
  const int mbase = blockIdx.y * 128;
  const int t    = threadIdx.x;
  const int w    = t >> 5;
  const int lane = t & 31;
  const int hi   = lane >> 4;
  const int ln   = lane & 15;
  const int wm   = (w & 3) * 32;   // wave M offset in block tile
  const int wn   = (w >> 2) * 64;  // wave N offset in block tile

  const int lrow = t >> 1;         // 0..127 : tile row this thread loads
  const int lcq  = (t & 1) << 4;   // 0 / 16 : K-half this thread loads

  v8f zero8;
#pragma unroll
  for (int r = 0; r < 8; ++r) zero8[r] = 0.0f;

  v8f acc[2][4];
#pragma unroll
  for (int i = 0; i < 2; ++i)
#pragma unroll
    for (int j = 0; j < 4; ++j) acc[i][j] = zero8;

  for (int kb = 0; kb < K; kb += 32) {
    // ---- stage A tile (128x32) into LDS as bf16 ----
    if constexpr (!A_BF16) {
      const float* ap = (const float*)Av + (size_t)(mbase + lrow) * K + kb + lcq;
      uint32_t* dst = (uint32_t*)(As + lrow * 40 + lcq);
#pragma unroll
      for (int jj = 0; jj < 4; ++jj) {
        float4 f = *(const float4*)(ap + jj * 4);
        dst[jj * 2 + 0] = pack_bf16x2(f.x, f.y);
        dst[jj * 2 + 1] = pack_bf16x2(f.z, f.w);
      }
      if (kb + 32 < K) __builtin_prefetch(ap + 32);
    } else {
      const uint16_t* ap =
          (const uint16_t*)Av + (size_t)(mbase + lrow) * K + kb + lcq;
      uint4* dst = (uint4*)(As + lrow * 40 + lcq);
      dst[0] = ((const uint4*)ap)[0];
      dst[1] = *(const uint4*)(ap + 8);
      if (kb + 32 < K) __builtin_prefetch(ap + 32);
    }
    // ---- stage B tile (128x32) into LDS as bf16 ----
    {
      const float* bp = B + (size_t)(nbase + lrow) * K + kb + lcq;
      uint32_t* dst = (uint32_t*)(Bs + lrow * 40 + lcq);
#pragma unroll
      for (int jj = 0; jj < 4; ++jj) {
        float4 f = *(const float4*)(bp + jj * 4);
        dst[jj * 2 + 0] = pack_bf16x2(f.x, f.y);
        dst[jj * 2 + 1] = pack_bf16x2(f.z, f.w);
      }
      if (kb + 32 < K) __builtin_prefetch(bp + 32);
    }
    __syncthreads();

    v16bf fa[2];
#pragma unroll
    for (int i = 0; i < 2; ++i) fa[i] = load_frag_a(As + (wm + i * 16) * 40, 40);
#pragma unroll
    for (int j = 0; j < 4; ++j) {
      v16bf fb = load_frag_b(Bs + (wn + j * 16) * 40, 40);
#pragma unroll
      for (int i = 0; i < 2; ++i) acc[i][j] = wmma_bf16(fa[i], fb, acc[i][j]);
    }
    __syncthreads();
  }

  // ---- epilogue ----
#pragma unroll
  for (int i = 0; i < 2; ++i) {
#pragma unroll
    for (int j = 0; j < 4; ++j) {
      const int mg = mbase + wm + i * 16 + hi * 8;  // row of acc element r = mg+r
      const int ng = nbase + wn + j * 16 + ln;      // column (lane)
      if constexpr (OUT_F32) {
        float* C = (float*)Cv;
#pragma unroll
        for (int r = 0; r < 8; ++r)
          C[(size_t)(mg + r) * N + ng] = acc[i][j][r];
      } else if constexpr (!TRANS_OUT) {
        uint16_t* C = (uint16_t*)Cv;
#pragma unroll
        for (int r = 0; r < 8; ++r)
          C[(size_t)(mg + r) * N + ng] = f32_to_bf16_rne(acc[i][j][r]);
      } else {  // bf16, transposed: C_t[N][M], rows pairwise-packed dword stores
        uint16_t* C = (uint16_t*)Cv;
#pragma unroll
        for (int r = 0; r < 8; r += 2)
          *(uint32_t*)(C + (size_t)ng * M + mg + r) =
              pack_bf16x2(acc[i][j][r], acc[i][j][r + 1]);
      }
    }
  }
}

// ---------------------------------------------------------------------------
// Flash-attention (causal) for one (batch, head, 128-row query tile).
//   Qp  : [B*S][DMODEL]  bf16 (head h occupies cols h*128..h*128+127)
//   Kp  : [B*S][KVDIM]   bf16 (group g occupies cols g*128..)
//   Vt  : [KVDIM][B*S]   bf16 (transposed projection: feature-major)
//   O   : [B*S][DMODEL]  bf16 (same head-major layout as Qp)
// 8 waves; wave w owns query rows [qtile*128 + 16w, +16). KV processed in
// 32-key blocks, double-buffered in LDS via async global->LDS B128 copies:
// block i+1 streams in (ASYNCcnt) while block i's 16 WMMAs execute; one
// s_wait_asynccnt 0 + one barrier per iteration.
// ---------------------------------------------------------------------------
__global__ __launch_bounds__(256) void gqa_attn_wmma(
    const uint16_t* __restrict__ Qp, const uint16_t* __restrict__ Kp,
    const uint16_t* __restrict__ Vt, uint16_t* __restrict__ O) {
  const float SCALE = 0.08838834764831845f;  // 1/sqrt(128)
  __shared__ __align__(16) uint16_t Ks[2][32 * 136];   // [key][feat], ld 136
  __shared__ __align__(16) uint16_t Vs[2][128 * 40];   // [feat][key], ld 40
  __shared__ __align__(16) uint16_t Ps[8 * 16 * 40];   // per-wave P scratch

  const int qtile = blockIdx.x;        // 0..15
  const int hg    = blockIdx.y;        // 0..31  global head
  const int b     = blockIdx.z;        // 0..1
  const int g     = hg >> 2;           // KV group
  const int t     = threadIdx.x;
  const int w     = t >> 5;
  const int lane  = t & 31;
  const int hi    = lane >> 4;
  const int ln    = lane & 15;
  const int qrow0 = qtile * 128 + w * 16;
  const int qmax  = qrow0 + 15;
  uint16_t* Pw = Ps + w * (16 * 40);

  // Per-thread tile-staging coordinates (whole 256-thread WG cooperates).
  const int kkey = t >> 3, kfq = (t & 7) << 4;   // K tile: 32 keys x 128 feat
  const int vf   = t >> 1, vkq = (t & 1) << 4;   // V tile: 128 feat x 32 keys

  // Issue async global->LDS copies for KV block at `kb` into buffer `buf`.
  auto issue_tile = [&](int kb, int buf) {
    const uint16_t* ksrc =
        Kp + (size_t)(b * SEQ + kb + kkey) * KVDIM + g * DHEAD + kfq;
    const uint32_t kdst = lds_off_of(&Ks[buf][kkey * 136 + kfq]);
    async_ld_b128(kdst, ksrc);
    async_ld_b128(kdst + 16, ksrc + 8);
    const uint16_t* vsrc =
        Vt + (size_t)(g * DHEAD + vf) * (BATCH * SEQ) + b * SEQ + kb + vkq;
    const uint32_t vdst = lds_off_of(&Vs[buf][vf * 40 + vkq]);
    async_ld_b128(vdst, vsrc);
    async_ld_b128(vdst + 16, vsrc + 8);
  };

  // Q fragments live in registers for the whole KV loop (4 chunks of K=32).
  v16bf qf[4];
#pragma unroll
  for (int c = 0; c < 4; ++c)
    qf[c] = load_frag_a(
        Qp + (size_t)(b * SEQ + qrow0) * DMODEL + hg * DHEAD + c * 32, DMODEL);

  v8f zero8;
#pragma unroll
  for (int r = 0; r < 8; ++r) zero8[r] = 0.0f;
  v8f accO[8];
#pragma unroll
  for (int f = 0; f < 8; ++f) accO[f] = zero8;
  float mrow[8], lrow[8];
#pragma unroll
  for (int r = 0; r < 8; ++r) { mrow[r] = -__builtin_inff(); lrow[r] = 0.0f; }

  const int kvend = qtile * 128 + 128;
  issue_tile(0, 0);

  int ib = 0;
  for (int kb = 0; kb < kvend; kb += 32, ib ^= 1) {
    wait_async0();      // this wave's copies for block kb have landed in LDS
    __syncthreads();    // all waves' copies visible; prev buffer free to reuse
    if (kb + 32 < kvend) issue_tile(kb + 32, ib ^ 1);  // overlap with compute

    if (kb <= qmax) {  // block contributes to this wave's rows (causal)
      const uint16_t* Kb = Ks[ib];
      const uint16_t* Vb = Vs[ib];
      // ---- scores: S(16x32) = Q(16x128) * K_blk(32x128)^T ----
      v8f s0 = zero8, s1 = zero8;
#pragma unroll
      for (int c = 0; c < 4; ++c) {
        v16bf k0 = load_frag_b(Kb + c * 32, 136);             // keys 0..15
        s0 = wmma_bf16(qf[c], k0, s0);
        v16bf k1 = load_frag_b(Kb + 16 * 136 + c * 32, 136);  // keys 16..31
        s1 = wmma_bf16(qf[c], k1, s1);
      }
      // ---- streaming softmax (f32), P -> bf16 via per-wave LDS scratch ----
#pragma unroll
      for (int r = 0; r < 8; ++r) {
        const int qi = qrow0 + r + hi * 8;
        float v0 = s0[r] * SCALE;
        float v1 = s1[r] * SCALE;
        if (kb + ln > qi)      v0 = -__builtin_inff();
        if (kb + 16 + ln > qi) v1 = -__builtin_inff();
        float mx = fmaxf(v0, v1);
        mx = fmaxf(mx, __shfl_xor(mx, 1, 32));
        mx = fmaxf(mx, __shfl_xor(mx, 2, 32));
        mx = fmaxf(mx, __shfl_xor(mx, 4, 32));
        mx = fmaxf(mx, __shfl_xor(mx, 8, 32));
        const float mnew = fmaxf(mrow[r], mx);   // finite: kb <= qrow0 <= qi
        const float corr = __expf(mrow[r] - mnew);
        mrow[r] = mnew;
        const float p0 = __expf(v0 - mnew);
        const float p1 = __expf(v1 - mnew);
        float ps = p0 + p1;
        ps += __shfl_xor(ps, 1, 32);
        ps += __shfl_xor(ps, 2, 32);
        ps += __shfl_xor(ps, 4, 32);
        ps += __shfl_xor(ps, 8, 32);
        lrow[r] = lrow[r] * corr + ps;
#pragma unroll
        for (int f = 0; f < 8; ++f) accO[f][r] *= corr;
        Pw[(r + hi * 8) * 40 + ln]      = f32_to_bf16_rne(p0);
        Pw[(r + hi * 8) * 40 + 16 + ln] = f32_to_bf16_rne(p1);
      }
      // ---- O(16x128) += P(16x32) * V_blk(32x128) ----
      v16bf pf = load_frag_a(Pw, 40);
#pragma unroll
      for (int f = 0; f < 8; ++f) {
        v16bf vb = load_frag_b(Vb + f * 16 * 40, 40);
        accO[f] = wmma_bf16(pf, vb, accO[f]);
      }
    }
  }

  // ---- normalize and store O (bf16, head-major) ----
#pragma unroll
  for (int f = 0; f < 8; ++f) {
#pragma unroll
    for (int r = 0; r < 8; ++r) {
      const float o = accO[f][r] / lrow[r];
      O[(size_t)(b * SEQ + qrow0 + r + hi * 8) * DMODEL + hg * DHEAD + f * 16 +
        ln] = f32_to_bf16_rne(o);
    }
  }
}

// ---------------------------------------------------------------------------
// Host-side launcher.
// Inputs: 0:q 1:k 2:v 3:w_q 4:w_k 5:w_v 6:w_out 7:mask (ignored; exact tril)
// Workspace: Qp bf16 [4096][4096] | Kp bf16 [4096][1024] |
//            Vt bf16 [1024][4096] | Ob bf16 [4096][4096]   (~84 MB)
// ---------------------------------------------------------------------------
extern "C" void kernel_launch(void* const* d_in, const int* in_sizes, int n_in,
                              void* d_out, int out_size, void* d_ws,
                              size_t ws_size, hipStream_t stream) {
  (void)in_sizes; (void)n_in; (void)out_size; (void)ws_size;
  const float* q     = (const float*)d_in[0];
  const float* k     = (const float*)d_in[1];
  const float* v     = (const float*)d_in[2];
  const float* w_q   = (const float*)d_in[3];
  const float* w_k   = (const float*)d_in[4];
  const float* w_v   = (const float*)d_in[5];
  const float* w_out = (const float*)d_in[6];

  const int M = BATCH * SEQ;  // 4096 rows
  uint16_t* Qp = (uint16_t*)d_ws;
  uint16_t* Kp = Qp + (size_t)M * DMODEL;
  uint16_t* Vt = Kp + (size_t)M * KVDIM;
  uint16_t* Ob = Vt + (size_t)KVDIM * M;
  float* out = (float*)d_out;

  dim3 blk(256);
  // Q projection: Qp = q @ w_q^T           [4096 x 4096]
  gemm_bf16_wmma<false, false, false>
      <<<dim3(DMODEL / 128, M / 128), blk, 0, stream>>>(q, w_q, Qp, M, DMODEL,
                                                        DMODEL);
  // K projection: Kp = k @ w_k^T           [4096 x 1024]
  gemm_bf16_wmma<false, false, false>
      <<<dim3(KVDIM / 128, M / 128), blk, 0, stream>>>(k, w_k, Kp, M, KVDIM,
                                                       DMODEL);
  // V projection (transposed out): Vt = (v @ w_v^T)^T   [1024 x 4096]
  gemm_bf16_wmma<false, false, true>
      <<<dim3(KVDIM / 128, M / 128), blk, 0, stream>>>(v, w_v, Vt, M, KVDIM,
                                                       DMODEL);
  // Causal GQA attention -> Ob             [4096 x 4096]
  gqa_attn_wmma<<<dim3(SEQ / 128, 32, BATCH), blk, 0, stream>>>(Qp, Kp, Vt, Ob);
  // Output projection: out = Ob @ w_out^T  (fp32 result)
  gemm_bf16_wmma<true, true, false>
      <<<dim3(DMODEL / 128, M / 128), blk, 0, stream>>>(Ob, w_out, out, M,
                                                        DMODEL, DMODEL);
}
